// MultiHeadAttention_37795712205068
// MI455X (gfx1250) — matrix operand order repaired and checked by reading, not compiled, not run on hardware
//
#include <hip/hip_runtime.h>
#include <hip/hip_bf16.h>

// ---------------- problem constants ----------------
constexpr int Nn  = 2;
constexpr int Tt  = 2048;
constexpr int Dd  = 1024;
constexpr int Hh  = 16;
constexpr int Gg  = 4;
constexpr int DQq = 64;
constexpr int DVv = 64;
constexpr float SOFTCAP = 30.0f;
constexpr float LN_EPS  = 1e-5f;
constexpr float ROPE_THETA = 10000.0f;

// d_out layout (floats), concatenated in reference return order
constexpr size_t OFF_OUT   = 0;                                   // N*T*D
constexpr size_t OFF_PROBS = OFF_OUT   + (size_t)Nn*Tt*Dd;        // N*H*T*T
constexpr size_t OFF_QMU   = OFF_PROBS + (size_t)Nn*Hh*Tt*Tt;     // N*T*G*H*DQ
constexpr size_t OFF_QLV   = OFF_QMU   + (size_t)Nn*Tt*Gg*Hh*DQq;
constexpr size_t OFF_KMU   = OFF_QLV   + (size_t)Nn*Tt*Gg*Hh*DQq; // N*T*H*DQ
constexpr size_t OFF_KLV   = OFF_KMU   + (size_t)Nn*Tt*Hh*DQq;
constexpr size_t OFF_VMU   = OFF_KLV   + (size_t)Nn*Tt*Hh*DQq;
constexpr size_t OFF_VLV   = OFF_VMU   + (size_t)Nn*Tt*Hh*DVv;

// workspace layout (bytes)
constexpr size_t WS_LNH   = 0;                       // ln(x) f16:  N*T*D
constexpr size_t WS_WQT   = WS_LNH + (size_t)Nn*Tt*Dd*2;            // Wq^T f16: 8192*1024
constexpr size_t WS_WKT   = WS_WQT + (size_t)8192*1024*2;           // Wk^T f16: 2048*1024
constexpr size_t WS_WVT   = WS_WKT + (size_t)2048*1024*2;           // Wv^T f16: 2048*1024
constexpr size_t WS_WOT   = WS_WVT + (size_t)2048*1024*2;           // Wo^T f16: 1024*1024
constexpr size_t WS_QEFF  = WS_WOT + (size_t)1024*1024*2;           // q_eff f16: N*H*T*DQ
constexpr size_t WS_KH    = WS_QEFF + (size_t)Nn*Hh*Tt*DQq*2;       // k rope f16: N*H*T*DQ
constexpr size_t WS_VTH   = WS_KH   + (size_t)Nn*Hh*Tt*DQq*2;       // v^T f16: N*H*DV*T
constexpr size_t WS_ATTN  = WS_VTH  + (size_t)Nn*Hh*DVv*Tt*2;       // attn out f16: N*T*H*DV

// ---------------- WMMA vector types ----------------
typedef __attribute__((ext_vector_type(16))) _Float16 v16h;
typedef __attribute__((ext_vector_type(8)))  _Float16 v8h;
typedef __attribute__((ext_vector_type(8)))  float    v8f;

__device__ __forceinline__ v16h pack16(v8h lo, v8h hi) {
  v16h r;
#pragma unroll
  for (int i = 0; i < 8; ++i) { r[i] = lo[i]; r[i + 8] = hi[i]; }
  return r;
}

__device__ __forceinline__ float wave_max(float v) {
#pragma unroll
  for (int o = 16; o > 0; o >>= 1) v = fmaxf(v, __shfl_xor(v, o));
  return v;
}
__device__ __forceinline__ float wave_sum(float v) {
#pragma unroll
  for (int o = 16; o > 0; o >>= 1) v += __shfl_xor(v, o);
  return v;
}

// ---------------- CDNA5 async DMA: global -> LDS (ASYNCcnt-tracked) ----------------
// 16B per lane, no VGPR data staging. LDS address = low 32 bits of generic
// pointer into __shared__ (LDS aperture truncation, ISA 10.2).
__device__ __forceinline__ void async_copy_b128(unsigned lds_off, const _Float16* gptr) {
  asm volatile("global_load_async_to_lds_b128 %0, %1, off"
               :: "v"(lds_off), "v"(gptr) : "memory");
}
__device__ __forceinline__ void wait_async_le2() {
  asm volatile("s_wait_asynccnt 2" ::: "memory");
}
__device__ __forceinline__ void wait_async_0() {
  asm volatile("s_wait_asynccnt 0" ::: "memory");
}
__device__ __forceinline__ unsigned lds_addr32(const void* p) {
  return (unsigned)(size_t)p;
}

// ---------------- hash-based Gaussian (reparam eps) ----------------
__device__ __forceinline__ unsigned hmix(unsigned x) {
  x ^= x >> 16; x *= 0x7feb352du;
  x ^= x >> 15; x *= 0x846ca68bu;
  x ^= x >> 16; return x;
}
__device__ __forceinline__ float rand_normal(unsigned seed, unsigned idx) {
  unsigned a = hmix(idx ^ (seed * 0x9E3779B9u));
  unsigned b = hmix(a + 0x85EBCA6Bu);
  float u1 = (float)((a >> 8) + 1u) * (1.0f / 16777216.0f);
  float u2 = (float)(b >> 8) * (1.0f / 16777216.0f);
  return sqrtf(-2.0f * __logf(u1)) * __cosf(6.28318530718f * u2);
}

// ---------------- kernel 1: LayerNorm -> f16 ----------------
__global__ void ln_kernel(const float* __restrict__ x, const float* __restrict__ gamma,
                          const float* __restrict__ beta, _Float16* __restrict__ out) {
  const int row = blockIdx.x;                 // N*T rows of D=1024
  const int tid = threadIdx.x;                // 256 threads, 4 elems each
  const float* xr = x + (size_t)row * Dd;
  float4 v = ((const float4*)xr)[tid];
  float s  = v.x + v.y + v.z + v.w;
  float s2 = v.x * v.x + v.y * v.y + v.z * v.z + v.w * v.w;
  __shared__ float red1[8], red2[8];
  s = wave_sum(s); s2 = wave_sum(s2);
  const int wid = tid >> 5, lane = tid & 31;
  if (lane == 0) { red1[wid] = s; red2[wid] = s2; }
  __syncthreads();
  if (tid < 32) {
    float a = (tid < 8) ? red1[tid] : 0.0f;
    float b = (tid < 8) ? red2[tid] : 0.0f;
    a = wave_sum(a); b = wave_sum(b);
    if (tid == 0) { red1[0] = a; red2[0] = b; }
  }
  __syncthreads();
  const float mean = red1[0] * (1.0f / Dd);
  const float var  = red2[0] * (1.0f / Dd) - mean * mean;
  const float inv  = rsqrtf(var + LN_EPS);
  const int base = tid * 4;
  float4 g = ((const float4*)gamma)[tid];
  float4 bb = ((const float4*)beta)[tid];
  _Float16* o = out + (size_t)row * Dd + base;
  o[0] = (_Float16)((v.x - mean) * inv * g.x + bb.x);
  o[1] = (_Float16)((v.y - mean) * inv * g.y + bb.y);
  o[2] = (_Float16)((v.z - mean) * inv * g.z + bb.z);
  o[3] = (_Float16)((v.w - mean) * inv * g.w + bb.w);
}

// ---------------- kernel 2: transpose + f32->f16 (weights) ----------------
__global__ void transpose_cvt_kernel(const float* __restrict__ W, _Float16* __restrict__ Wt,
                                     int K, int Ncol) {
  long long idx = (long long)blockIdx.x * 256 + threadIdx.x;   // over Ncol*K, Wt[j*K+k]
  if (idx >= (long long)K * Ncol) return;
  int j = (int)(idx / K), k = (int)(idx % K);
  Wt[idx] = (_Float16)W[(size_t)k * Ncol + j];
}

// ---------------- kernel 3: generic WMMA GEMM, async double-buffered LDS ----------------
// C[M,Ncol] = A[M,K](f16,rowmaj) * Bt[Ncol,K](f16,rowmaj == B colmaj) + bias
// split>0: col<split -> out0[row*split+col], else out1[row*split+col-split]
__global__ void gemm_wmma_kernel(const _Float16* __restrict__ A, const _Float16* __restrict__ Bt,
                                 const float* __restrict__ bias, float* __restrict__ out0,
                                 float* __restrict__ out1, int M, int K, int Ncol, int split) {
  __shared__ __align__(32) _Float16 As[2][64 * 32];
  __shared__ __align__(32) _Float16 Bs[2][64 * 32];
  const int tid = threadIdx.x;
  const int w = tid >> 5, lane = tid & 31;
  const int n0 = blockIdx.x * 64, m0 = blockIdx.y * 64;
  const int arow = lane & 15, ahalf = lane >> 4;
  const int wm = (w & 3) * 16, wn = (w >> 2) * 32;
  const int lr = tid >> 2, lc = (tid & 3) * 8;     // LDS staging: 64 rows x 32 k, 16B/lane

  // per-thread DMA endpoints
  const _Float16* gA = &A[(size_t)(m0 + lr) * K + lc];
  const _Float16* gB = &Bt[(size_t)(n0 + lr) * K + lc];
  const unsigned ldsA[2] = { lds_addr32(&As[0][lr * 32 + lc]), lds_addr32(&As[1][lr * 32 + lc]) };
  const unsigned ldsB[2] = { lds_addr32(&Bs[0][lr * 32 + lc]), lds_addr32(&Bs[1][lr * 32 + lc]) };

  v8f acc[2] = {};

  // prologue: DMA tile 0 into buffer 0
  async_copy_b128(ldsA[0], gA);
  async_copy_b128(ldsB[0], gB);

  int cur = 0;
  for (int kt = 0; kt < K; kt += 32) {
    if (kt + 32 < K) {
      // DMA next tile into the other buffer while we compute this one
      async_copy_b128(ldsA[cur ^ 1], gA + kt + 32);
      async_copy_b128(ldsB[cur ^ 1], gB + kt + 32);
      wait_async_le2();                 // the 2 older copies (current tile) are done
    } else {
      wait_async_0();
    }
    __syncthreads();                    // current tile visible to all waves

    const _Float16* ap = &As[cur][(wm + arow) * 32 + ahalf * 16];
    v16h afrag = pack16(*(const v8h*)ap, *(const v8h*)(ap + 8));
#pragma unroll
    for (int ct = 0; ct < 2; ++ct) {
      const _Float16* bp = &Bs[cur][(wn + ct * 16 + arow) * 32 + ahalf * 16];
      v16h bfrag = *(const v16h*)bp;
      acc[ct] = __builtin_amdgcn_wmma_f32_16x16x32_f16(false, afrag, false, bfrag,
                                                       (short)0, acc[ct], false, false);
    }
    __syncthreads();                    // reads done before this buffer is re-DMA'd
    cur ^= 1;
  }

#pragma unroll
  for (int ct = 0; ct < 2; ++ct) {
#pragma unroll
    for (int r = 0; r < 8; ++r) {
      int row = m0 + wm + r + 8 * ahalf;
      int col = n0 + wn + ct * 16 + arow;
      float val = acc[ct][r] + bias[col];
      if (split > 0) {
        if (col < split) out0[(size_t)row * split + col] = val;
        else             out1[(size_t)row * split + (col - split)] = val;
      } else {
        out0[(size_t)row * Ncol + col] = val;
      }
    }
  }
}

// ---------------- kernel 4a: reparam + RoPE + sum over G (Q) ----------------
__global__ void reparam_rope_q_kernel(const float* __restrict__ qmu, const float* __restrict__ qlv,
                                      _Float16* __restrict__ qeff) {
  int idx = blockIdx.x * 256 + threadIdx.x;   // N*H*T*DQ
  int d = idx & (DQq - 1);
  int r = idx >> 6;
  int t = r % Tt;
  int h = (r / Tt) & (Hh - 1);
  int n = r / (Tt * Hh);
  float inv = __powf(ROPE_THETA, -(float)(d & ~1) * (1.0f / DQq));
  float sn, cs; __sincosf((float)t * inv, &sn, &cs);
  const int dp = d ^ 1;
  float acc = 0.0f;
#pragma unroll
  for (int g = 0; g < Gg; ++g) {
    size_t rowb = ((size_t)n * Tt + t) * (Gg * Hh * DQq);
    int c  = (g * Hh + h) * DQq + d;
    int cp = (g * Hh + h) * DQq + dp;
    float x  = qmu[rowb + c]  + rand_normal(42u, (unsigned)(rowb + c))  * __expf(0.5f * qlv[rowb + c]);
    float xp = qmu[rowb + cp] + rand_normal(42u, (unsigned)(rowb + cp)) * __expf(0.5f * qlv[rowb + cp]);
    float rot = (d & 1) ? xp : -xp;
    acc += x * cs + rot * sn;
  }
  // fold 1/sqrt(DQ) score scale into q
  qeff[(((size_t)n * Hh + h) * Tt + t) * DQq + d] = (_Float16)(acc * 0.125f);
}

// ---------------- kernel 4b: reparam + RoPE (K) ----------------
__global__ void reparam_rope_k_kernel(const float* __restrict__ kmu, const float* __restrict__ klv,
                                      _Float16* __restrict__ kh) {
  int idx = blockIdx.x * 256 + threadIdx.x;   // N*H*T*DQ
  int d = idx & (DQq - 1);
  int r = idx >> 6;
  int t = r % Tt;
  int h = (r / Tt) & (Hh - 1);
  int n = r / (Tt * Hh);
  float inv = __powf(ROPE_THETA, -(float)(d & ~1) * (1.0f / DQq));
  float sn, cs; __sincosf((float)t * inv, &sn, &cs);
  size_t rowb = ((size_t)n * Tt + t) * (Hh * DQq);
  int c  = h * DQq + d;
  int cp = h * DQq + (d ^ 1);
  float x  = kmu[rowb + c]  + rand_normal(43u, (unsigned)(rowb + c))  * __expf(0.5f * klv[rowb + c]);
  float xp = kmu[rowb + cp] + rand_normal(43u, (unsigned)(rowb + cp)) * __expf(0.5f * klv[rowb + cp]);
  float rot = (d & 1) ? xp : -xp;
  kh[(((size_t)n * Hh + h) * Tt + t) * DQq + d] = (_Float16)(x * cs + rot * sn);
}

// ---------------- kernel 4c: reparam (V), store transposed [n,h,dv,s] ----------------
__global__ void reparam_v_kernel(const float* __restrict__ vmu, const float* __restrict__ vlv,
                                 _Float16* __restrict__ vt) {
  int idx = blockIdx.x * 256 + threadIdx.x;   // N*H*T*DV
  int d = idx & (DVv - 1);
  int r = idx >> 6;
  int t = r % Tt;
  int h = (r / Tt) & (Hh - 1);
  int n = r / (Tt * Hh);
  size_t rowb = ((size_t)n * Tt + t) * (Hh * DVv);
  int c = h * DVv + d;
  float x = vmu[rowb + c] + rand_normal(44u, (unsigned)(rowb + c)) * __expf(0.5f * vlv[rowb + c]);
  vt[(((size_t)n * Hh + h) * DVv + d) * Tt + t] = (_Float16)x;
}

// ---------------- kernel 5: WMMA QK^T + softcap + mask + softmax -> probs ----------------
// one workgroup = one (n,h,16-row tile); 16x2048 f32 score strip lives in LDS (128KB)
__global__ void attn_scores_kernel(const _Float16* __restrict__ qeff, const _Float16* __restrict__ kh,
                                   const int* __restrict__ lengths, float* __restrict__ probs) {
  extern __shared__ float sbuf[];             // [16][Tt]
  const int ttile = blockIdx.x, h = blockIdx.y, n = blockIdx.z;
  const int tid = threadIdx.x;
  const int w = tid >> 5, lane = tid & 31;
  const int arow = lane & 15, ahalf = lane >> 4;
  const int t0 = ttile * 16;
  const int len = lengths[n];

  // Q fragments (16 rows x 64 d), loaded once per wave
  const size_t qbase = (((size_t)n * Hh + h) * Tt + t0) * DQq;
  v16h afrag[2];
#pragma unroll
  for (int kk = 0; kk < 2; ++kk) {
    const _Float16* p = qeff + qbase + (size_t)arow * DQq + kk * 32 + ahalf * 16;
    afrag[kk] = pack16(*(const v8h*)p, *(const v8h*)(p + 8));
  }

  const size_t kbase = ((size_t)n * Hh + h) * (size_t)Tt * DQq;
  for (int st = 0; st < 16; ++st) {           // wave w covers s in [w*256, w*256+256)
    const int s0 = w * 256 + st * 16;
    v8f acc = {};
#pragma unroll
    for (int kk = 0; kk < 2; ++kk) {
      const _Float16* bp = kh + kbase + (size_t)(s0 + arow) * DQq + kk * 32 + ahalf * 16;
      v16h bfrag = *(const v16h*)bp;
      acc = __builtin_amdgcn_wmma_f32_16x16x32_f16(false, afrag[kk], false, bfrag,
                                                   (short)0, acc, false, false);
    }
#pragma unroll
    for (int r = 0; r < 8; ++r) {
      const int row = r + 8 * ahalf;          // 0..15
      const int tg = t0 + row;
      const int sg = s0 + arow;
      float sc = acc[r];
      sc = SOFTCAP * tanhf(sc * (1.0f / SOFTCAP));
      if (sg > tg || sg >= len) sc = -__builtin_inff();
      sbuf[row * Tt + sg] = sc;
    }
  }
  __syncthreads();

  // softmax: wave w owns rows 2w, 2w+1; full 2048-wide scan by 32 lanes
  for (int rr = 0; rr < 2; ++rr) {
    const int row = w * 2 + rr;
    float m = -__builtin_inff();
    for (int s = lane; s < Tt; s += 32) m = fmaxf(m, sbuf[row * Tt + s]);
    m = wave_max(m);
    float sum = 0.0f;
    for (int s = lane; s < Tt; s += 32) {
      float sc = sbuf[row * Tt + s];
      float e = (sc > -1e30f) ? __expf(sc - m) : 0.0f;
      sbuf[row * Tt + s] = e;
      sum += e;
    }
    sum = wave_sum(sum);
    const float invs = (sum > 0.0f) ? (1.0f / sum) : 0.0f;
    float* prow = probs + (((size_t)n * Hh + h) * Tt + (t0 + row)) * Tt;
    for (int s = lane; s < Tt; s += 32) prow[s] = sbuf[row * Tt + s] * invs;
  }
}

// ---------------- kernel 6: WMMA PV (probs f32 -> f16 on the fly) ----------------
__global__ void pv_wmma_kernel(const float* __restrict__ probs, const _Float16* __restrict__ vt,
                               _Float16* __restrict__ attn_h) {
  const int blk = blockIdx.x, h = blockIdx.y, n = blockIdx.z;
  const int tid = threadIdx.x;
  const int w = tid >> 5, lane = tid & 31;
  const int arow = lane & 15, ahalf = lane >> 4;
  const int t0 = blk * 128 + w * 16;          // each wave owns a distinct 16-row tile
  const float* prow = probs + (((size_t)n * Hh + h) * Tt + (t0 + arow)) * Tt;
  const _Float16* vbase = vt + ((size_t)n * Hh + h) * (size_t)DVv * Tt;
  v8f acc[4] = {};
  for (int s0 = 0; s0 < Tt; s0 += 32) {
    const float* p = prow + s0 + ahalf * 16;
    if (s0 + 32 < Tt) __builtin_prefetch(p + 32, 0, 3);   // stream 512MB probs re-read
    float4 f0 = *(const float4*)p;
    float4 f1 = *(const float4*)(p + 4);
    float4 f2 = *(const float4*)(p + 8);
    float4 f3 = *(const float4*)(p + 12);
    v16h a;
    a[0]=(_Float16)f0.x; a[1]=(_Float16)f0.y; a[2]=(_Float16)f0.z; a[3]=(_Float16)f0.w;
    a[4]=(_Float16)f1.x; a[5]=(_Float16)f1.y; a[6]=(_Float16)f1.z; a[7]=(_Float16)f1.w;
    a[8]=(_Float16)f2.x; a[9]=(_Float16)f2.y; a[10]=(_Float16)f2.z; a[11]=(_Float16)f2.w;
    a[12]=(_Float16)f3.x; a[13]=(_Float16)f3.y; a[14]=(_Float16)f3.z; a[15]=(_Float16)f3.w;
#pragma unroll
    for (int ct = 0; ct < 4; ++ct) {
      const _Float16* bp = vbase + (size_t)(ct * 16 + arow) * Tt + s0 + ahalf * 16;
      v16h bfrag = *(const v16h*)bp;
      acc[ct] = __builtin_amdgcn_wmma_f32_16x16x32_f16(false, a, false, bfrag,
                                                       (short)0, acc[ct], false, false);
    }
  }
#pragma unroll
  for (int ct = 0; ct < 4; ++ct) {
#pragma unroll
    for (int r = 0; r < 8; ++r) {
      const int trow = t0 + r + 8 * ahalf;
      const int dv = ct * 16 + arow;
      attn_h[((size_t)n * Tt + trow) * (Hh * DVv) + h * DVv + dv] = (_Float16)acc[ct][r];
    }
  }
}

// ---------------- launcher ----------------
extern "C" void kernel_launch(void* const* d_in, const int* in_sizes, int n_in,
                              void* d_out, int out_size, void* d_ws, size_t ws_size,
                              hipStream_t stream) {
  (void)in_sizes; (void)n_in; (void)out_size; (void)ws_size;
  const float* x       = (const float*)d_in[0];   // query_sequences (== key == value)
  const int*   lengths = (const int*)d_in[3];
  const float* gamma   = (const float*)d_in[6];
  const float* beta    = (const float*)d_in[7];
  const float* Wq = (const float*)d_in[8];   const float* bq = (const float*)d_in[9];
  const float* Wk = (const float*)d_in[10];  const float* bk = (const float*)d_in[11];
  const float* Wv = (const float*)d_in[12];  const float* bv = (const float*)d_in[13];
  const float* Wo = (const float*)d_in[14];  const float* bo = (const float*)d_in[15];

  float* out = (float*)d_out;
  char*  ws  = (char*)d_ws;
  _Float16* ln_h   = (_Float16*)(ws + WS_LNH);
  _Float16* Wq_t   = (_Float16*)(ws + WS_WQT);
  _Float16* Wk_t   = (_Float16*)(ws + WS_WKT);
  _Float16* Wv_t   = (_Float16*)(ws + WS_WVT);
  _Float16* Wo_t   = (_Float16*)(ws + WS_WOT);
  _Float16* qeff_h = (_Float16*)(ws + WS_QEFF);
  _Float16* k_h    = (_Float16*)(ws + WS_KH);
  _Float16* v_t_h  = (_Float16*)(ws + WS_VTH);
  _Float16* attn_h = (_Float16*)(ws + WS_ATTN);

  const int M = Nn * Tt;       // 4096 rows

  // 1) LayerNorm -> f16
  ln_kernel<<<M, 256, 0, stream>>>(x, gamma, beta, ln_h);

  // 2) weight transpose + f16 convert
  {
    long long tq = (long long)Dd * (Gg * Hh * 2 * DQq);
    long long tk = (long long)Dd * (Hh * 2 * DQq);
    long long to = (long long)(Hh * DVv) * Dd;
    transpose_cvt_kernel<<<(unsigned)((tq + 255) / 256), 256, 0, stream>>>(Wq, Wq_t, Dd, Gg * Hh * 2 * DQq);
    transpose_cvt_kernel<<<(unsigned)((tk + 255) / 256), 256, 0, stream>>>(Wk, Wk_t, Dd, Hh * 2 * DQq);
    transpose_cvt_kernel<<<(unsigned)((tk + 255) / 256), 256, 0, stream>>>(Wv, Wv_t, Dd, Hh * 2 * DVv);
    transpose_cvt_kernel<<<(unsigned)((to + 255) / 256), 256, 0, stream>>>(Wo, Wo_t, Hh * DVv, Dd);
  }

  // 3) projections (WMMA + async LDS DMA), epilogue splits mu/logvar straight into d_out
  gemm_wmma_kernel<<<dim3((Gg*Hh*2*DQq)/64, M/64), 256, 0, stream>>>(
      ln_h, Wq_t, bq, out + OFF_QMU, out + OFF_QLV, M, Dd, Gg*Hh*2*DQq, Gg*Hh*DQq);
  gemm_wmma_kernel<<<dim3((Hh*2*DQq)/64, M/64), 256, 0, stream>>>(
      ln_h, Wk_t, bk, out + OFF_KMU, out + OFF_KLV, M, Dd, Hh*2*DQq, Hh*DQq);
  gemm_wmma_kernel<<<dim3((Hh*2*DVv)/64, M/64), 256, 0, stream>>>(
      ln_h, Wv_t, bv, out + OFF_VMU, out + OFF_VLV, M, Dd, Hh*2*DVv, Hh*DVv);

  // 4) reparam + RoPE (+ G-sum for Q), f16 fragments for attention
  {
    const int tot = Nn * Hh * Tt * DQq;      // == for V with DVv
    reparam_rope_q_kernel<<<tot / 256, 256, 0, stream>>>(out + OFF_QMU, out + OFF_QLV, qeff_h);
    reparam_rope_k_kernel<<<tot / 256, 256, 0, stream>>>(out + OFF_KMU, out + OFF_KLV, k_h);
    reparam_v_kernel<<<tot / 256, 256, 0, stream>>>(out + OFF_VMU, out + OFF_VLV, v_t_h);
  }

  // 5) WMMA QK^T + softcap + masks + softmax -> probs (16x2048 f32 strip in 128KB LDS)
  attn_scores_kernel<<<dim3(Tt / 16, Hh, Nn), 256, 16 * Tt * sizeof(float), stream>>>(
      qeff_h, k_h, lengths, out + OFF_PROBS);

  // 6) WMMA PV -> attn_h (f16)
  pv_wmma_kernel<<<dim3(Tt / 128, Hh, Nn), 256, 0, stream>>>(out + OFF_PROBS, v_t_h, attn_h);

  // 7) output projection (WMMA) -> out
  gemm_wmma_kernel<<<dim3(Dd / 64, M / 64), 256, 0, stream>>>(
      attn_h, Wo_t, bo, out + OFF_OUT, nullptr, M, Hh * DVv, Dd, 0);
}
